// RelativeMultiHeadAttention_34505767256972
// MI455X (gfx1250) — matrix operand order repaired and checked by reading, not compiled, not run on hardware
//
#include <hip/hip_runtime.h>
#include <hip/hip_bf16.h>

// Relative multi-head attention for MI455X (gfx1250, wave32, WMMA).
// All GEMMs run on v_wmma_f32_16x16x32_bf16 (fp32->bf16 inputs, f32 accum).
// Attention is flash-style (two-pass softmax per 16-query tile) so the
// [B,H,S,S] score tensor (256 MB) never touches HBM.
// Dense projections use 2x2 register tiling: 4 WMMAs per (2A + 2B) fragment
// fetch, doubling operand reuse so the WMMA pipes aren't feed-limited.

#define B_ 8
#define S_ 1024
#define DM 512
#define H_ 8
#define HD 64
#define NR 33   // 2*MAX_REL+1
#define SW 8    // waves per block (8 x wave32 = 256 threads)

typedef __bf16 bf16_t;
typedef bf16_t v16bf __attribute__((ext_vector_type(16)));
typedef float  v8f   __attribute__((ext_vector_type(8)));

union FragU { v16bf v; uint4 q[2]; };

__device__ __forceinline__ unsigned short f2bfu(float f) {
  unsigned u = __builtin_bit_cast(unsigned, f);
  unsigned r = u + 0x7FFFu + ((u >> 16) & 1u);   // round-to-nearest-even
  return (unsigned short)(r >> 16);
}

// A-matrix fragment (16x32 bf16): lane L holds row M=L&15; K = 16*hi..16*hi+15
// contiguous (hi = L>>4). Two b128 loads.
__device__ __forceinline__ v16bf load_fragA(const unsigned short* row_kb, int hi) {
  FragU f;
  f.q[0] = *(const uint4*)(row_kb + 16 * hi);
  f.q[1] = *(const uint4*)(row_kb + 16 * hi + 8);
  return f.v;
}
// B-matrix fragment (32x16 bf16): lane L holds col N=L&15; K = 16*hi..16*hi+15
// contiguous -> one 32B chunk (two b128 loads).
__device__ __forceinline__ v16bf load_fragB(const unsigned short* col_kb, int hi) {
  FragU f;
  const uint4* p = (const uint4*)(col_kb + 16 * hi);
  f.q[0] = p[0];
  f.q[1] = p[1];
  return f.v;
}
__device__ __forceinline__ v8f wmma_bf16(v16bf a, v16bf b, v8f c) {
  return __builtin_amdgcn_wmma_f32_16x16x32_bf16(false, a, false, b, (short)0, c,
                                                 false, false);
}

// ---------------------------------------------------------------- fp32 -> bf16
// 8 elements per thread: two b128 loads -> one packed b128 store.
__global__ void cvt_bf16_kernel(const float* __restrict__ in,
                                unsigned short* __restrict__ out, int nVec8) {
  int i = blockIdx.x * blockDim.x + threadIdx.x;
  if (i < nVec8) {
    const float4* p = (const float4*)in + (size_t)i * 2;
    float4 f0 = p[0], f1 = p[1];
    uint4 o;
    o.x = (unsigned)f2bfu(f0.x) | ((unsigned)f2bfu(f0.y) << 16);
    o.y = (unsigned)f2bfu(f0.z) | ((unsigned)f2bfu(f0.w) << 16);
    o.z = (unsigned)f2bfu(f1.x) | ((unsigned)f2bfu(f1.y) << 16);
    o.w = (unsigned)f2bfu(f1.z) | ((unsigned)f2bfu(f1.w) << 16);
    ((uint4*)out)[i] = o;
  }
}

// ------------------------------------------------- Q/K/V projection + head split
// out[i,j] = sum_d X[i,d]*W[j,d] + bias[j]; W row-major IS the B-operand layout.
// 2x2 register tiling: each wave owns a 32x32 output block (4 WMMA tiles).
// transpose=0: out[b,h,s,dh] row layout (Q,K);  transpose=1: out[b,h,dh,s] (V^T).
__global__ __launch_bounds__(256) void proj_qkv_kernel(
    const unsigned short* __restrict__ X, const unsigned short* __restrict__ W,
    const float* __restrict__ bias, unsigned short* __restrict__ out,
    int transpose) {
  const int lane = threadIdx.x & 31, wave = threadIdx.x >> 5;
  const int hi = lane >> 4, n = lane & 15;
  const int tile = blockIdx.x * SW + wave;   // 4096 tiles: 256 Mb x 16 Nb
  const int mb = tile >> 4, nb = tile & 15;  // 32x32 block
  const unsigned short* arow0 = X + (size_t)(mb * 32 + n) * DM;
  const unsigned short* arow1 = arow0 + (size_t)16 * DM;
  const unsigned short* bcol0 = W + (size_t)(nb * 32 + n) * DM;
  const unsigned short* bcol1 = bcol0 + (size_t)16 * DM;
  v8f c00 = {}, c01 = {}, c10 = {}, c11 = {};
#pragma unroll 2
  for (int kk = 0; kk < DM / 32; ++kk) {
    v16bf a0 = load_fragA(arow0 + kk * 32, hi);
    v16bf a1 = load_fragA(arow1 + kk * 32, hi);
    v16bf b0 = load_fragB(bcol0 + kk * 32, hi);
    v16bf b1 = load_fragB(bcol1 + kk * 32, hi);
    c00 = wmma_bf16(a0, b0, c00);
    c01 = wmma_bf16(a0, b1, c01);
    c10 = wmma_bf16(a1, b0, c10);
    c11 = wmma_bf16(a1, b1, c11);
  }
  const v8f* cs[4] = {&c00, &c01, &c10, &c11};
#pragma unroll
  for (int t = 0; t < 4; ++t) {
    const v8f& c = *cs[t];
    const int mt16 = mb * 32 + (t >> 1) * 16;
    const int j = nb * 32 + (t & 1) * 16 + n;
    const float bj = bias[j];
    const int h = j >> 6, dh = j & (HD - 1);
#pragma unroll
    for (int e = 0; e < 8; ++e) {  // C layout: M = e + 8*hi, N = lane&15
      int mg = mt16 + e + 8 * hi;
      int b = mg >> 10, s = mg & (S_ - 1);
      size_t idx = transpose ? (((size_t)(b * H_ + h)) * HD + dh) * S_ + s
                             : (((size_t)(b * H_ + h)) * S_ + s) * HD + dh;
      out[idx] = f2bfu(c[e] + bj);
    }
  }
}

// --------------------------------------------------------- fused rel-attention
// One wave32 owns one 16-query tile of one (b,h). Two-pass softmax; probs are
// routed C-layout -> LDS -> A-layout for the attn@V WMMAs.
__global__ __launch_bounds__(256) void attn_kernel(
    const unsigned short* __restrict__ Q, const unsigned short* __restrict__ K,
    const unsigned short* __restrict__ Vt, const unsigned short* __restrict__ relk,
    const float* __restrict__ relv, unsigned short* __restrict__ ctx) {
  __shared__ float s_relv[NR * HD];
  __shared__ float s_p[SW][16][36];                        // P = Q @ relk^T
  __shared__ float s_sb[SW][16][34];                       // bucket sums
  __shared__ __align__(16) unsigned short s_pt[SW][16][32];// prob tile (bf16)

  // Stage rel_v via the CDNA5 async global->LDS path when available
  // (ASYNCcnt-tracked, CDNA5 ISA); else plain loads.
#if defined(__gfx1250__) && __has_builtin(__builtin_amdgcn_global_load_async_to_lds_b32) && __has_builtin(__builtin_amdgcn_s_wait_asynccnt)
  for (int i = threadIdx.x; i < NR * HD; i += 256) {
    __builtin_amdgcn_global_load_async_to_lds_b32(
        (__attribute__((address_space(1))) int*)(relv + i),
        (__attribute__((address_space(3))) int*)(&s_relv[i]), 0, 0);
  }
  __builtin_amdgcn_s_wait_asynccnt(0);
#else
  for (int i = threadIdx.x; i < NR * HD; i += 256) s_relv[i] = relv[i];
#endif
  __syncthreads();

  const int lane = threadIdx.x & 31, wave = threadIdx.x >> 5;
  const int hi = lane >> 4, n = lane & 15;
  const int tile = blockIdx.x * SW + wave;  // 4096 tiles = B*H*64
  const int qt = tile & 63;
  const int bh = tile >> 6;

  const unsigned short* Qb = Q + (size_t)bh * S_ * HD;
  const unsigned short* Kb = K + (size_t)bh * S_ * HD;
  const unsigned short* Vb = Vt + (size_t)bh * HD * S_;

  const unsigned short* qrow = Qb + (size_t)(qt * 16 + n) * HD;
  v16bf a0 = load_fragA(qrow, hi);        // Q dims 0..31
  v16bf a1 = load_fragA(qrow + 32, hi);   // Q dims 32..63

  float (*pP)[36] = s_p[wave];
  float (*pSB)[34] = s_sb[wave];
  unsigned short (*pPT)[32] = s_pt[wave];

  // P[q][r] = Q[q] . rel_k[r]   (3 column tiles cover r=0..47, clamp >32)
#pragma unroll
  for (int cb = 0; cb < 3; ++cb) {
    int r = cb * 16 + n;
    int rc = r > 32 ? 32 : r;
    const unsigned short* rrow = relk + rc * HD;
    v8f c = {};
    c = wmma_bf16(a0, load_fragB(rrow, hi), c);
    c = wmma_bf16(a1, load_fragB(rrow + 32, hi), c);
#pragma unroll
    for (int e = 0; e < 8; ++e)
      if (r <= 32) pP[e + 8 * hi][r] = c[e];
  }
  for (int i = lane; i < 16 * 34; i += 32) (&pSB[0][0])[i] = 0.f;

  const float rscale = 0.35355339059327373f;  // 1/sqrt(B=8) per reference
  const int qg0 = qt * 16;

  // ---- pass 1: row max over all 1024 keys
  float mx[8];
#pragma unroll
  for (int e = 0; e < 8; ++e) mx[e] = -3.0e38f;
  for (int kt = 0; kt < 64; ++kt) {
    const unsigned short* krow = Kb + (size_t)(kt * 16 + n) * HD;
    if (kt + 1 < 64) __builtin_prefetch(krow + 16 * HD, 0, 1);  // next K tile
    v8f c = {};
    c = wmma_bf16(a0, load_fragB(krow, hi), c);
    c = wmma_bf16(a1, load_fragB(krow + 32, hi), c);
    int kg = kt * 16 + n;
#pragma unroll
    for (int e = 0; e < 8; ++e) {
      int m = e + 8 * hi;
      int d = kg - (qg0 + m);
      d = d < -16 ? -16 : (d > 16 ? 16 : d);
      float sc = (c[e] + pP[m][d + 16]) * rscale;
      mx[e] = fmaxf(mx[e], sc);
    }
  }
#pragma unroll
  for (int e = 0; e < 8; ++e)
#pragma unroll
    for (int off = 8; off; off >>= 1)
      mx[e] = fmaxf(mx[e], __shfl_xor(mx[e], off, 16));

  // ---- pass 2: probabilities, bucket sums, attn @ V
  float l[8], s0[8], s32[8];
#pragma unroll
  for (int e = 0; e < 8; ++e) { l[e] = 0.f; s0[e] = 0.f; s32[e] = 0.f; }
  v8f o0 = {}, o1 = {}, o2 = {}, o3 = {};
  for (int kt = 0; kt < 64; ++kt) {
    const unsigned short* krow = Kb + (size_t)(kt * 16 + n) * HD;
    v8f c = {};
    c = wmma_bf16(a0, load_fragB(krow, hi), c);
    c = wmma_bf16(a1, load_fragB(krow + 32, hi), c);
    int kg = kt * 16 + n;
#pragma unroll
    for (int e = 0; e < 8; ++e) {
      int m = e + 8 * hi;
      int d = kg - (qg0 + m);
      int bkt = (d < -16 ? -16 : (d > 16 ? 16 : d)) + 16;
      float sc = (c[e] + pP[m][bkt]) * rscale;
      float p = __expf(sc - mx[e]);
      l[e] += p;
      if (bkt == 0) s0[e] += p;          // many k share edge buckets:
      else if (bkt == 32) s32[e] += p;   //   register-accumulate, reduce later
      else pSB[m][bkt] = p;              // middle bucket: unique (q,k) writer
      pPT[m][(kt & 1) * 16 + n] = f2bfu(p);  // C-layout scalar store to LDS
    }
    if (kt & 1) {  // consume 32 keys: A-frag from LDS, B-frags from V^T
      int kb = (kt - 1) * 16;
      v16bf pa = load_fragA(&pPT[n][0], hi);
      o0 = wmma_bf16(pa, load_fragB(Vb + (size_t)(0 * 16 + n) * S_ + kb, hi), o0);
      o1 = wmma_bf16(pa, load_fragB(Vb + (size_t)(1 * 16 + n) * S_ + kb, hi), o1);
      o2 = wmma_bf16(pa, load_fragB(Vb + (size_t)(2 * 16 + n) * S_ + kb, hi), o2);
      o3 = wmma_bf16(pa, load_fragB(Vb + (size_t)(3 * 16 + n) * S_ + kb, hi), o3);
    }
  }
#pragma unroll
  for (int e = 0; e < 8; ++e) {
#pragma unroll
    for (int off = 8; off; off >>= 1) {
      l[e] += __shfl_xor(l[e], off, 16);
      s0[e] += __shfl_xor(s0[e], off, 16);
      s32[e] += __shfl_xor(s32[e], off, 16);
    }
    if (n == 0) {  // one writer per row per half-wave
      int m = e + 8 * hi;
      pSB[m][0] = s0[e];
      pSB[m][32] = s32[e];
    }
  }

  // ---- epilogue: ctx = (attn@V + sB @ rel_v) / l
  const int h = bh & (H_ - 1), b = bh >> 3;
#pragma unroll
  for (int t = 0; t < 4; ++t) {
    const v8f& oc = (t == 0 ? o0 : t == 1 ? o1 : t == 2 ? o2 : o3);
#pragma unroll
    for (int e = 0; e < 8; ++e) {
      int m = e + 8 * hi;
      int d = t * 16 + n;
      float w2 = 0.f;
      for (int r = 0; r < NR; ++r) w2 += pSB[m][r] * s_relv[r * HD + d];
      float val = (oc[e] + w2) / l[e];
      size_t idx = ((size_t)(b * S_ + qg0 + m)) * DM + h * HD + d;
      ctx[idx] = f2bfu(val);
    }
  }
}

// ------------------------------------------------------------ output projection
// Same 2x2 register tiling; fp32 output straight to d_out.
__global__ __launch_bounds__(256) void out_proj_kernel(
    const unsigned short* __restrict__ X, const unsigned short* __restrict__ W,
    const float* __restrict__ bias, float* __restrict__ out) {
  const int lane = threadIdx.x & 31, wave = threadIdx.x >> 5;
  const int hi = lane >> 4, n = lane & 15;
  const int tile = blockIdx.x * SW + wave;
  const int mb = tile >> 4, nb = tile & 15;
  const unsigned short* arow0 = X + (size_t)(mb * 32 + n) * DM;
  const unsigned short* arow1 = arow0 + (size_t)16 * DM;
  const unsigned short* bcol0 = W + (size_t)(nb * 32 + n) * DM;
  const unsigned short* bcol1 = bcol0 + (size_t)16 * DM;
  v8f c00 = {}, c01 = {}, c10 = {}, c11 = {};
#pragma unroll 2
  for (int kk = 0; kk < DM / 32; ++kk) {
    v16bf a0 = load_fragA(arow0 + kk * 32, hi);
    v16bf a1 = load_fragA(arow1 + kk * 32, hi);
    v16bf b0 = load_fragB(bcol0 + kk * 32, hi);
    v16bf b1 = load_fragB(bcol1 + kk * 32, hi);
    c00 = wmma_bf16(a0, b0, c00);
    c01 = wmma_bf16(a0, b1, c01);
    c10 = wmma_bf16(a1, b0, c10);
    c11 = wmma_bf16(a1, b1, c11);
  }
  const v8f* cs[4] = {&c00, &c01, &c10, &c11};
#pragma unroll
  for (int t = 0; t < 4; ++t) {
    const v8f& c = *cs[t];
    const int mt16 = mb * 32 + (t >> 1) * 16;
    const int j = nb * 32 + (t & 1) * 16 + n;
    const float bj = bias[j];
#pragma unroll
    for (int e = 0; e < 8; ++e) {
      int mg = mt16 + e + 8 * hi;
      out[(size_t)mg * DM + j] = c[e] + bj;
    }
  }
}

extern "C" void kernel_launch(void* const* d_in, const int* in_sizes, int n_in,
                              void* d_out, int out_size, void* d_ws, size_t ws_size,
                              hipStream_t stream) {
  (void)in_sizes; (void)n_in; (void)out_size; (void)ws_size;
  const float* query = (const float*)d_in[0];
  const float* key = (const float*)d_in[1];
  const float* value = (const float*)d_in[2];
  const float* Wq = (const float*)d_in[3];
  const float* bq = (const float*)d_in[4];
  const float* Wk = (const float*)d_in[5];
  const float* bk = (const float*)d_in[6];
  const float* Wv = (const float*)d_in[7];
  const float* bv = (const float*)d_in[8];
  const float* Wo = (const float*)d_in[9];
  const float* bo = (const float*)d_in[10];
  const float* relk_f = (const float*)d_in[11];
  const float* relv_f = (const float*)d_in[12];

  char* ws = (char*)d_ws;
  size_t off = 0;
  auto take = [&](size_t bytes) {
    char* p = ws + off;
    off = (off + bytes + 255) & ~(size_t)255;
    return p;
  };
  const size_t nX = (size_t)B_ * S_ * DM;  // 4,194,304
  const size_t nW = (size_t)DM * DM;       // 262,144
  unsigned short* xq = (unsigned short*)take(nX * 2);
  unsigned short* xk = (unsigned short*)take(nX * 2);
  unsigned short* xv = (unsigned short*)take(nX * 2);
  unsigned short* wq = (unsigned short*)take(nW * 2);
  unsigned short* wk = (unsigned short*)take(nW * 2);
  unsigned short* wv = (unsigned short*)take(nW * 2);
  unsigned short* wo = (unsigned short*)take(nW * 2);
  unsigned short* qh = (unsigned short*)take(nX * 2);  // [B,H,S,64]
  unsigned short* kh = (unsigned short*)take(nX * 2);  // [B,H,S,64]
  unsigned short* vt = (unsigned short*)take(nX * 2);  // [B,H,64,S]
  unsigned short* relkb = (unsigned short*)take((size_t)NR * HD * 2);
  unsigned short* ctxb = (unsigned short*)take(nX * 2);  // [B*S, 512]

  // fp32 -> bf16 staging, 8 elements/thread (bandwidth-trivial overall)
  const int vX = (int)(nX / 8), vW = (int)(nW / 8), vR = NR * HD / 8;
  cvt_bf16_kernel<<<(vX + 255) / 256, 256, 0, stream>>>(query, xq, vX);
  cvt_bf16_kernel<<<(vX + 255) / 256, 256, 0, stream>>>(key, xk, vX);
  cvt_bf16_kernel<<<(vX + 255) / 256, 256, 0, stream>>>(value, xv, vX);
  cvt_bf16_kernel<<<(vW + 255) / 256, 256, 0, stream>>>(Wq, wq, vW);
  cvt_bf16_kernel<<<(vW + 255) / 256, 256, 0, stream>>>(Wk, wk, vW);
  cvt_bf16_kernel<<<(vW + 255) / 256, 256, 0, stream>>>(Wv, wv, vW);
  cvt_bf16_kernel<<<(vW + 255) / 256, 256, 0, stream>>>(Wo, wo, vW);
  cvt_bf16_kernel<<<(vR + 255) / 256, 256, 0, stream>>>(relk_f, relkb, vR);

  // projections: 256 Mb x 16 Nb = 4096 32x32 blocks / 8 waves = 512 blocks
  proj_qkv_kernel<<<512, 256, 0, stream>>>(xq, wq, bq, qh, 0);
  proj_qkv_kernel<<<512, 256, 0, stream>>>(xk, wk, bk, kh, 0);
  proj_qkv_kernel<<<512, 256, 0, stream>>>(xv, wv, bv, vt, 1);

  // fused attention: B*H*(S/16) = 4096 query tiles / 8 waves = 512 blocks
  attn_kernel<<<512, 256, 0, stream>>>(qh, kh, vt, relkb, relv_f, ctxb);

  // output projection -> fp32 d_out
  out_proj_kernel<<<512, 256, 0, stream>>>(ctxb, wo, bo, (float*)d_out);
}
